// DCGRUCell_47141561041224
// MI455X (gfx1250) — matrix operand order repaired and checked by reading, not compiled, not run on hardware
//
#include <hip/hip_runtime.h>

// ---------------- problem constants (match reference) ----------------
#define NNODES 50000
#define DIN    2
#define UNITS  64
#define FEATS  128              // D*B
#define MMATS  7                // num_matrices
#define ROWS   896              // FEATS * MMATS  (output rows)
#define KPAD   50048            // NNODES rounded up to multiple of 64
#define PCOLS  128              // 64 u-columns + 64 c-columns (r is dead)
#define KSTEPS (KPAD / 32)      // 1564 (pack grid along K)
#define DSLABS (KPAD / 64)      // 782  (64-wide K slabs in GEMM)
#define SPLITK 112
#define SLABS_PER_SPLIT 7       // ceil(782/112)
#define LDA 72                  // padded LDS row stride (shorts) -> conflict-free b128
#define LDB 72

typedef __attribute__((ext_vector_type(16))) __bf16       v16bf;
typedef __attribute__((ext_vector_type(8)))  float        v8f;
typedef __attribute__((ext_vector_type(4)))  unsigned int u32x4;

union FragBF { v16bf v; u32x4 u[2]; };
union Cvt16  { unsigned short s[16]; u32x4 u[2]; };

__device__ __forceinline__ unsigned short f2bf(float x) {
  unsigned int u = __float_as_uint(x);
  u += 0x7FFFu + ((u >> 16) & 1u);   // round-to-nearest-even
  return (unsigned short)(u >> 16);
}

// ---------------- x0 = transpose(inputs,(1,2,0)).reshape(N,128) ------
__global__ void build_x0(const float* __restrict__ inputs, float* __restrict__ X0) {
  int t = blockIdx.x * blockDim.x + threadIdx.x;   // exactly N*128 threads
  int n = t >> 7;
  int f = t & 127;
  int d = f >> 6;
  int b = f & 63;
  X0[t] = inputs[((size_t)b * NNODES + n) * DIN + d];
}

// ---------------- dst = -src (seed for x2 = 2*A*x1 - xc) -------------
__global__ void seed_neg(const float4* __restrict__ src, float4* __restrict__ dst) {
  int t = blockIdx.x * blockDim.x + threadIdx.x;   // exactly plane/4 threads
  float4 v = src[t];
  dst[t] = make_float4(-v.x, -v.y, -v.z, -v.w);
}

// ---------------- COO SpMM: Xdst[rows] += scale*vals*Xsrc[cols] ------
// one wave32 per nonzero; each lane covers 4 of the 128 features
__global__ void spmm_atomic(const int* __restrict__ rows, const int* __restrict__ cols,
                            const float* __restrict__ vals,
                            const float* __restrict__ Xsrc, float* __restrict__ Xdst,
                            float scale, int nnz) {
  int gw   = (int)((blockIdx.x * blockDim.x + threadIdx.x) >> 5);
  int lane = threadIdx.x & 31;
  if (gw >= nnz) return;
  int   r = rows[gw];
  int   c = cols[gw];
  float v = vals[gw] * scale;
  float4 x = ((const float4*)(Xsrc + (size_t)c * FEATS))[lane];
  float* dst = Xdst + (size_t)r * FEATS + lane * 4;
  atomicAdd(dst + 0, v * x.x);
  atomicAdd(dst + 1, v * x.y);
  atomicAdd(dst + 2, v * x.z);
  atomicAdd(dst + 3, v * x.w);
}

// ---------------- pack X (m,n,f) -> bf16 A[(f*7+m)][n], K zero-padded
__global__ void pack_a(const float* __restrict__ X, unsigned short* __restrict__ A) {
  __shared__ float tile[32][129];
  const int m  = blockIdx.y;
  const int n0 = blockIdx.x * 32;
  const float* Xm = X + (size_t)m * NNODES * FEATS;
  for (int i = threadIdx.x; i < 32 * FEATS; i += 256) {
    int nn = i >> 7, f = i & 127;
    int n = n0 + nn;
    tile[nn][f] = (n < NNODES) ? Xm[(size_t)n * FEATS + f] : 0.0f;
  }
  __syncthreads();
  int f  = threadIdx.x >> 1;
  int ns = (threadIdx.x & 1) * 16;
  Cvt16 cv;
#pragma unroll
  for (int i = 0; i < 16; ++i) cv.s[i] = f2bf(tile[ns + i][f]);
  u32x4* dst = (u32x4*)(A + (size_t)(f * MMATS + m) * KPAD + n0 + ns);
  dst[0] = cv.u[0];
  dst[1] = cv.u[1];
}

// ---- pack weights -> bf16 WT[j][k]; j<64: w_ru col 64+j (u), else w_c col j-64
__global__ void pack_w(const float* __restrict__ w_ru, const float* __restrict__ w_c,
                       unsigned short* __restrict__ WT) {
  __shared__ float tile[32][129];
  const int k0 = blockIdx.x * 32;
  for (int i = threadIdx.x; i < 32 * 128; i += 256) {
    int kk = i >> 7, j = i & 127;
    int k = k0 + kk;
    float v = 0.0f;
    if (k < NNODES) v = (j < 64) ? w_ru[(size_t)k * 128 + 64 + j]
                                 : w_c[(size_t)k * 64 + (j - 64)];
    tile[kk][j] = v;
  }
  __syncthreads();
  int j  = threadIdx.x >> 1;
  int ks = (threadIdx.x & 1) * 16;
  Cvt16 cv;
#pragma unroll
  for (int i = 0; i < 16; ++i) cv.s[i] = f2bf(tile[ks + i][j]);
  u32x4* dst = (u32x4*)(WT + (size_t)j * KPAD + k0 + ks);
  dst[0] = cv.u[0];
  dst[1] = cv.u[1];
}

// ---------------- LDS-blocked split-K bf16 WMMA GEMM -----------------
// block = 8 waves -> 64 rows x 128 cols; per 64-K slab stage A(8KB)+B(16KB)
// in LDS; each wave computes a 32x32 sub-tile = 4 accumulators, 8 wmma/slab.
__global__ void __launch_bounds__(256) gemm_wmma(const unsigned short* __restrict__ A,
                                                 const unsigned short* __restrict__ WT,
                                                 float* __restrict__ P) {
  __shared__ unsigned short sA[64 * LDA];    //  9.0 KB
  __shared__ unsigned short sB[128 * LDB];   // 18.0 KB
  const int tid  = threadIdx.x;
  const int lane = tid & 31;
  const int wv   = tid >> 5;
  const int row_base = blockIdx.x * 64;      // 14 row blocks * 64 = 896
  int slab_begin = blockIdx.z * SLABS_PER_SPLIT;
  int slab_end   = slab_begin + SLABS_PER_SPLIT;
  if (slab_end > DSLABS) slab_end = DSLABS;

  // global->LDS staging assignments
  const int a_row = tid >> 2;                // 0..63
  const int a_q   = (tid & 3) * 16;          // 32B per thread
  const unsigned short* gA = A + (size_t)(row_base + a_row) * KPAD + a_q;
  unsigned short*       lA = sA + a_row * LDA + a_q;

  const int b_col = tid >> 1;                // 0..127
  const int b_h   = (tid & 1) * 32;          // 64B per thread
  const unsigned short* gB = WT + (size_t)b_col * KPAD + b_h;
  unsigned short*       lB = sB + b_col * LDB + b_h;

  // per-wave 32x32 sub-tile
  const int wr   = (wv & 1) * 32;            // row group
  const int wc   = (wv >> 1) * 32;           // col group
  const int half = lane >> 4;                // 0 or 1
  const int l15  = lane & 15;
  const int ka   = half * 8;                 // A: lanes16-31 -> K 8..15 / 24..31
  const int kb   = half * 8;                 // B: lanes16-31 -> K 8..15 / 24..31
  const unsigned short* fA0 = sA + (wr + l15) * LDA + ka;
  const unsigned short* fA1 = fA0 + 16 * LDA;
  const unsigned short* fB0 = sB + (wc + l15) * LDB + kb;
  const unsigned short* fB1 = fB0 + 16 * LDB;

  v8f acc00 = {}, acc01 = {}, acc10 = {}, acc11 = {};

  for (int s = slab_begin; s < slab_end; ++s) {
    const int k0 = s * 64;
    // stage the slab
    *(u32x4*)(lA)      = *(const u32x4*)(gA + k0);
    *(u32x4*)(lA + 8)  = *(const u32x4*)(gA + k0 + 8);
    *(u32x4*)(lB)      = *(const u32x4*)(gB + k0);
    *(u32x4*)(lB + 8)  = *(const u32x4*)(gB + k0 + 8);
    *(u32x4*)(lB + 16) = *(const u32x4*)(gB + k0 + 16);
    *(u32x4*)(lB + 24) = *(const u32x4*)(gB + k0 + 24);
    __builtin_prefetch(gA + k0 + 64, 0, 1);
    __builtin_prefetch(gB + k0 + 64, 0, 1);
    __syncthreads();
#pragma unroll
    for (int ks = 0; ks < 64; ks += 32) {
      FragBF a0, a1, b0, b1;
      a0.u[0] = *(const u32x4*)(fA0 + ks);
      a0.u[1] = *(const u32x4*)(fA0 + ks + 16);
      a1.u[0] = *(const u32x4*)(fA1 + ks);
      a1.u[1] = *(const u32x4*)(fA1 + ks + 16);
      b0.u[0] = *(const u32x4*)(fB0 + ks);
      b0.u[1] = *(const u32x4*)(fB0 + ks + 16);
      b1.u[0] = *(const u32x4*)(fB1 + ks);
      b1.u[1] = *(const u32x4*)(fB1 + ks + 16);
      acc00 = __builtin_amdgcn_wmma_f32_16x16x32_bf16(false, a0.v, false, b0.v,
                                                      (short)0, acc00, false, false);
      acc01 = __builtin_amdgcn_wmma_f32_16x16x32_bf16(false, a0.v, false, b1.v,
                                                      (short)0, acc01, false, false);
      acc10 = __builtin_amdgcn_wmma_f32_16x16x32_bf16(false, a1.v, false, b0.v,
                                                      (short)0, acc10, false, false);
      acc11 = __builtin_amdgcn_wmma_f32_16x16x32_bf16(false, a1.v, false, b1.v,
                                                      (short)0, acc11, false, false);
    }
    __syncthreads();
  }

  // C/D layout: lane l -> N = l&15; VGPR i -> M = (l>>4)*8 + i
  float* pr0 = P + (size_t)(row_base + wr + half * 8) * PCOLS + wc + l15;
  float* pr1 = pr0 + (size_t)16 * PCOLS;
#pragma unroll
  for (int i = 0; i < 8; ++i) {
    atomicAdd(pr0 + (size_t)i * PCOLS,      acc00[i]);
    atomicAdd(pr0 + (size_t)i * PCOLS + 16, acc01[i]);
    atomicAdd(pr1 + (size_t)i * PCOLS,      acc10[i]);
    atomicAdd(pr1 + (size_t)i * PCOLS + 16, acc11[i]);
  }
}

// ---------------- out = u*hx + (1-u)*tanh(c) -------------------------
__global__ void epilogue(const float* __restrict__ P, const float* __restrict__ b_ru,
                         const float* __restrict__ b_c, const float* __restrict__ hx,
                         float* __restrict__ out) {
  int t = blockIdx.x * blockDim.x + threadIdx.x;
  if (t >= ROWS * UNITS) return;
  int r = t >> 6;
  int j = t & 63;
  float ul = P[(size_t)r * PCOLS + j] + b_ru[64 + j];
  float cl = P[(size_t)r * PCOLS + 64 + j] + b_c[j];
  float u  = 1.0f / (1.0f + __expf(-ul));
  float c  = tanhf(cl);
  out[t] = u * hx[t] + (1.0f - u) * c;
}

extern "C" void kernel_launch(void* const* d_in, const int* in_sizes, int n_in,
                              void* d_out, int out_size, void* d_ws, size_t ws_size,
                              hipStream_t stream) {
  const float* inputs   = (const float*)d_in[0];
  const float* hx       = (const float*)d_in[1];
  const float* w_ru     = (const float*)d_in[2];
  const float* b_ru     = (const float*)d_in[3];
  const float* w_c      = (const float*)d_in[4];
  const float* b_c      = (const float*)d_in[5];
  const float* lap_vals = (const float*)d_in[6];
  const float* rw_vals  = (const float*)d_in[7];
  const int*   lap_rows = (const int*)d_in[8];
  const int*   lap_cols = (const int*)d_in[9];
  const int*   rw_rows  = (const int*)d_in[10];
  const int*   rw_cols  = (const int*)d_in[11];
  const int nnz = in_sizes[6];

  // workspace layout
  char* ws = (char*)d_ws;
  float* X = (float*)ws;                                         // 7 * N * 128 f32
  const size_t plane  = (size_t)NNODES * FEATS;                  // floats per plane
  const size_t xbytes = (size_t)MMATS * plane * sizeof(float);
  unsigned short* A  = (unsigned short*)(ws + xbytes);           // 896 x KPAD bf16
  const size_t abytes = (size_t)ROWS * KPAD * sizeof(unsigned short);
  unsigned short* WT = (unsigned short*)(ws + xbytes + abytes);  // 128 x KPAD bf16
  const size_t wtbytes = (size_t)PCOLS * KPAD * sizeof(unsigned short);
  float* P = (float*)(ws + xbytes + abytes + wtbytes);           // 896 x 128 f32
  (void)ws_size; (void)n_in; (void)out_size;

  // zero the pure-accumulation planes (x1 of each support) and P
  hipMemsetAsync(X + 1 * plane, 0, plane * sizeof(float), stream);
  hipMemsetAsync(X + 3 * plane, 0, plane * sizeof(float), stream);
  hipMemsetAsync(X + 5 * plane, 0, plane * sizeof(float), stream);
  hipMemsetAsync(P, 0, (size_t)ROWS * PCOLS * sizeof(float), stream);

  // plane 0 = x0
  build_x0<<<(NNODES * FEATS) / 256, 256, 0, stream>>>(inputs, X);

  const int spmm_blocks = (nnz + 7) / 8;        // 8 waves (nnz) per 256-thread block
  const int seed_blocks = (int)(plane / 4 / 256);

  // support 0 (scaled laplacian):  x1a = L x0 ; x2a = 2 L x1a - x0
  spmm_atomic<<<spmm_blocks, 256, 0, stream>>>(lap_rows, lap_cols, lap_vals,
                                               X + 0 * plane, X + 1 * plane, 1.0f, nnz);
  seed_neg<<<seed_blocks, 256, 0, stream>>>((const float4*)(X + 0 * plane),
                                            (float4*)(X + 2 * plane));
  spmm_atomic<<<spmm_blocks, 256, 0, stream>>>(lap_rows, lap_cols, lap_vals,
                                               X + 1 * plane, X + 2 * plane, 2.0f, nnz);
  // support 1 (rw^T): x1b = R x1a ; x2b = 2 R x1b - x1a
  spmm_atomic<<<spmm_blocks, 256, 0, stream>>>(rw_rows, rw_cols, rw_vals,
                                               X + 1 * plane, X + 3 * plane, 1.0f, nnz);
  seed_neg<<<seed_blocks, 256, 0, stream>>>((const float4*)(X + 1 * plane),
                                            (float4*)(X + 4 * plane));
  spmm_atomic<<<spmm_blocks, 256, 0, stream>>>(rw_rows, rw_cols, rw_vals,
                                               X + 3 * plane, X + 4 * plane, 2.0f, nnz);
  // support 2 (rw^T): x1c = R x1b ; x2c = 2 R x1c - x1b
  spmm_atomic<<<spmm_blocks, 256, 0, stream>>>(rw_rows, rw_cols, rw_vals,
                                               X + 3 * plane, X + 5 * plane, 1.0f, nnz);
  seed_neg<<<seed_blocks, 256, 0, stream>>>((const float4*)(X + 3 * plane),
                                            (float4*)(X + 6 * plane));
  spmm_atomic<<<spmm_blocks, 256, 0, stream>>>(rw_rows, rw_cols, rw_vals,
                                               X + 5 * plane, X + 6 * plane, 2.0f, nnz);

  // pack to bf16 WMMA layouts (K padded to 50048 with zeros)
  pack_a<<<dim3(KSTEPS, MMATS), 256, 0, stream>>>(X, A);
  pack_w<<<KSTEPS, 256, 0, stream>>>(w_ru, w_c, WT);

  // LDS-blocked split-K WMMA GEMM: 14 row blocks x 112 K-slices, 8 waves each
  gemm_wmma<<<dim3(ROWS / 64, 1, SPLITK), 256, 0, stream>>>(A, WT, P);

  // gate + combine
  epilogue<<<(ROWS * UNITS + 255) / 256, 256, 0, stream>>>(P, b_ru, b_c, hx,
                                                           (float*)d_out);
}
